// WaveNet_43310450213095
// MI455X (gfx1250) — matrix operand order repaired and checked by reading, not compiled, not run on hardware
//
#include <hip/hip_runtime.h>
#include <math.h>
#include <stdint.h>

typedef __bf16 v16bf __attribute__((ext_vector_type(16)));
typedef __bf16 v8bf  __attribute__((ext_vector_type(8)));
typedef float  v8f   __attribute__((ext_vector_type(8)));

#define T_LEN 8192
#define ROWS  16384
#define RES   128
#define GATE  256
#define SKIPC 128
#define OUTC  256
#define CPAD  96
#define SQRT_HALF  0.70710678118654752f
#define SKIP_SCALE 0.22360679774997896f   /* sqrt(1/20) */

static __device__ __forceinline__ __bf16 f2bf(float f) { return (__bf16)f; }
static __device__ __forceinline__ float  bf2f(__bf16 b) { return (float)b; }

static __device__ __forceinline__ v8f wmma_bf16(v16bf a, v16bf b, v8f c) {
  return __builtin_amdgcn_wmma_f32_16x16x32_bf16(false, a, false, b, (short)0, c,
                                                 false, false);
}

// Async global->LDS copy of one 16B chunk per lane (ASYNCcnt path).
// GV addressing form: 64-bit global address in a VGPR pair, SADDR = off.
static __device__ __forceinline__ void async_b128(unsigned lds_off,
                                                  const void* gaddr) {
  asm volatile("global_load_async_to_lds_b128 %0, %1, off"
               :
               : "v"(lds_off), "v"((unsigned long long)(uintptr_t)gaddr)
               : "memory");
}
static __device__ __forceinline__ void wait_async0() {
  asm volatile("s_wait_asynccnt 0x0" ::: "memory");
}

// A-fragment (16x32 bf16) for this lane from a bf16 row-major matrix.
// rowptr = this lane's row start; cb = kb*32 + (lane>=16 ? 8 : 0).
static __device__ __forceinline__ v16bf load_afrag(const __bf16* rowptr, int cb) {
  v8bf lo = *(const v8bf*)(rowptr + cb);
  v8bf hi = *(const v8bf*)(rowptr + cb + 16);
  v16bf a;
#pragma unroll
  for (int i = 0; i < 8; ++i) { a[i] = lo[i]; a[8 + i] = hi[i]; }
  return a;
}

// ---------------------------------------------------------------- prep kernels

// Swizzle (nLayers, Kdim, Ndim) f32 weights into bf16 WMMA B-fragments:
// dst[layer][kb][ntile][lane][e]  (fragment = 32 lanes x 16 bf16 = 1KB;
// one K-stage = numNt contiguous fragments).
__global__ void wn_swz(const float* __restrict__ src, __bf16* __restrict__ dst,
                       int Kdim, int numKb, int Ndim, long total) {
  long idx = (long)blockIdx.x * blockDim.x + threadIdx.x;
  if (idx >= total) return;
  int e    = (int)(idx & 15);
  int lane = (int)((idx >> 4) & 31);
  long f   = idx >> 9;
  int numNt = Ndim >> 4;
  int nt = (int)(f % numNt); f /= numNt;
  int kb = (int)(f % numKb);
  long layer = f / numKb;
  int K = kb * 32 + ((lane >= 16) ? 8 : 0) + ((e < 8) ? e : e + 8);
  int N = nt * 16 + (lane & 15);
  float v = (K < Kdim) ? src[layer * (long)Kdim * Ndim + (long)K * Ndim + N] : 0.f;
  dst[idx] = f2bf(v);
}

__global__ void wn_cpad(const float* __restrict__ c, __bf16* __restrict__ dst,
                        long total) {
  long idx = (long)blockIdx.x * blockDim.x + threadIdx.x;
  if (idx >= total) return;
  long row = idx / CPAD;
  int ch = (int)(idx % CPAD);
  dst[idx] = f2bf((ch < 80) ? c[row * 80 + ch] : 0.f);
}

__global__ void wn_zero(float* __restrict__ p, long n) {
  long i = (long)blockIdx.x * blockDim.x + threadIdx.x;
  if (i < n) p[i] = 0.f;
}

// ------------------------------------------------------------- compute kernels

// h = x @ w_first + b_first  (x: 16384x256 f32, h: 16384x128 bf16)
__global__ __launch_bounds__(256) void wn_first(const float* __restrict__ x,
                                                const __bf16* __restrict__ wf,
                                                const float* __restrict__ bf,
                                                __bf16* __restrict__ hout) {
  const int lane = threadIdx.x & 31;
  const int wid  = threadIdx.x >> 5;
  const int t0   = (blockIdx.x * 8 + wid) * 16;
  const int m    = lane & 15;
  const bool hiL = lane >= 16;
  const int chn  = m;

  const float* xrow = x + (size_t)(t0 + m) * OUTC;
  v16bf af[8];
#pragma unroll
  for (int kb = 0; kb < 8; ++kb) {
    const int cb = kb * 32 + (hiL ? 8 : 0);
    v16bf a;
#pragma unroll
    for (int i = 0; i < 8; ++i) {
      a[i]     = f2bf(xrow[cb + i]);
      a[8 + i] = f2bf(xrow[cb + 16 + i]);
    }
    af[kb] = a;
  }
#pragma unroll
  for (int n = 0; n < 8; ++n) {
    const float b0 = bf[n * 16 + chn];
    v8f acc;
#pragma unroll
    for (int j = 0; j < 8; ++j) acc[j] = b0;
#pragma unroll
    for (int kb = 0; kb < 8; ++kb) {
      v16bf b = *(const v16bf*)(wf + ((size_t)(kb * 8 + n) * 32 + lane) * 16);
      acc = wmma_bf16(af[kb], b, acc);
    }
#pragma unroll
    for (int j = 0; j < 8; ++j)
      hout[(size_t)(t0 + j + (hiL ? 8 : 0)) * RES + n * 16 + chn] = f2bf(acc[j]);
  }
}

// One residual layer: gate = dilated conv + conditioning, gated activation,
// skip accumulation, residual update. Weights staged to LDS via async copies.
__global__ __launch_bounds__(256) void wn_layer(
    const __bf16* __restrict__ hin, __bf16* __restrict__ hout,
    const __bf16* __restrict__ cpad, const __bf16* __restrict__ wdil,
    const __bf16* __restrict__ wc, const __bf16* __restrict__ wskip,
    const __bf16* __restrict__ wres, const float* __restrict__ bdil,
    const float* __restrict__ bskip, const float* __restrict__ bres,
    float* __restrict__ skips, int dil) {
  __shared__ __bf16 wsh[2 * 16 * 512];   // 32KB weight staging (2 x 16KB)
  __shared__ __bf16 zbuf[8 * 16 * 128];  // 32KB z staging (per-wave 4KB)
  const int tid  = threadIdx.x;
  const int lane = tid & 31;
  const int wid  = tid >> 5;
  const int t0   = (blockIdx.x * 8 + wid) * 16;
  const int m    = lane & 15;
  const bool hiL = lane >= 16;
  const int chn  = m;
  __bf16* zw = zbuf + wid * 16 * 128;
  const unsigned wsh_base = (unsigned)(uintptr_t)(&wsh[0]);

  // A-fragment loader for gate stage s (0..11 = dilated taps, 12..14 = cond).
  auto loadA = [&](int s) -> v16bf {
    v16bf a;
#pragma unroll
    for (int i = 0; i < 16; ++i) a[i] = (__bf16)0.f;
    if (s < 12) {
      const int tap = s >> 2, kb = s & 3;
      const int off = (2 - tap) * dil;
      if ((t0 % T_LEN) + m - off >= 0)
        a = load_afrag(hin + (size_t)(t0 - off + m) * RES, kb * 32 + (hiL ? 8 : 0));
    } else {
      a = load_afrag(cpad + (size_t)(t0 + m) * CPAD, (s - 12) * 32 + (hiL ? 8 : 0));
    }
    return a;
  };
  // Stage one 16KB weight chunk (16 fragments) into LDS half `half`.
  auto copyStage = [&](int s, int half) {
    const __bf16* src =
        (s < 12) ? (wdil + (size_t)s * 8192) : (wc + (size_t)(s - 12) * 8192);
    const char* g = (const char*)src + (size_t)tid * 64;
    const unsigned lb = wsh_base + (unsigned)half * 16384 + (unsigned)tid * 64;
#pragma unroll
    for (int k = 0; k < 4; ++k) async_b128(lb + k * 16, g + k * 16);
  };

  v8f acc[16];
#pragma unroll
  for (int n = 0; n < 16; ++n)
#pragma unroll
    for (int j = 0; j < 8; ++j) acc[n][j] = 0.f;

  // ---- gate: 15 K-stages, double-buffered weight staging + A prefetch
  copyStage(0, 0);
  v16bf a_cur = loadA(0);
  wait_async0();
  __syncthreads();
  for (int s = 0; s < 15; ++s) {
    const __bf16* wcur = wsh + (s & 1) * 8192 + lane * 16;
    v16bf a_nxt = a_cur;
    if (s + 1 < 15) {
      copyStage(s + 1, (s + 1) & 1);  // overlaps this stage's WMMA burst
      a_nxt = loadA(s + 1);
    }
#pragma unroll
    for (int n = 0; n < 16; ++n) {
      v16bf b = *(const v16bf*)(wcur + n * 512);
      acc[n] = wmma_bf16(a_cur, b, acc[n]);
    }
    wait_async0();
    __syncthreads();
    a_cur = a_nxt;
  }

  // ---- stage skip weights (32 frags = 32KB) while doing activation VALU work
  {
    const char* g = (const char*)wskip + (size_t)tid * 128;
    const unsigned lb = wsh_base + (unsigned)tid * 128;
#pragma unroll
    for (int k = 0; k < 8; ++k) async_b128(lb + k * 16, g + k * 16);
  }

  // ---- gated activation z = tanh(a+ba)*sigmoid(b+bb); stage z in LDS bf16
#pragma unroll
  for (int n = 0; n < 8; ++n) {
    const float ba = bdil[n * 16 + chn];
    const float bb = bdil[128 + n * 16 + chn];
#pragma unroll
    for (int j = 0; j < 8; ++j) {
      const float av = acc[n][j] + ba;
      const float bv = acc[n + 8][j] + bb;
      const float z  = tanhf(av) * (1.0f / (1.0f + expf(-bv)));
      zw[(j + (hiL ? 8 : 0)) * 128 + n * 16 + chn] = f2bf(z);
    }
  }
  // re-read z as WMMA A fragments (wave-local LDS, no barrier needed)
  v16bf zf[4];
#pragma unroll
  for (int kb = 0; kb < 4; ++kb) {
    const int cb = kb * 32 + (hiL ? 8 : 0);
    v8bf lo = *(const v8bf*)(zw + m * 128 + cb);
    v8bf hb = *(const v8bf*)(zw + m * 128 + cb + 16);
    v16bf a;
#pragma unroll
    for (int i = 0; i < 8; ++i) { a[i] = lo[i]; a[8 + i] = hb[i]; }
    zf[kb] = a;
  }
  // preload skip accumulator (overlaps the async weight copy)
  v8f sacc[8];
#pragma unroll
  for (int n = 0; n < 8; ++n) {
    const float bs = bskip[n * 16 + chn];
#pragma unroll
    for (int j = 0; j < 8; ++j)
      sacc[n][j] =
          skips[(size_t)(t0 + j + (hiL ? 8 : 0)) * SKIPC + n * 16 + chn] + bs;
  }
  wait_async0();
  __syncthreads();

  // ---- skips += z @ w_skip + b_skip  (B frags from LDS)
#pragma unroll
  for (int n = 0; n < 8; ++n) {
#pragma unroll
    for (int kb = 0; kb < 4; ++kb) {
      v16bf b = *(const v16bf*)(wsh + (size_t)(kb * 8 + n) * 512 + lane * 16);
      sacc[n] = wmma_bf16(zf[kb], b, sacc[n]);
    }
#pragma unroll
    for (int j = 0; j < 8; ++j)
      skips[(size_t)(t0 + j + (hiL ? 8 : 0)) * SKIPC + n * 16 + chn] = sacc[n][j];
  }
  __syncthreads();  // everyone done reading skip weights

  // ---- stage res weights; overlap with residual C-operand loads
  {
    const char* g = (const char*)wres + (size_t)tid * 128;
    const unsigned lb = wsh_base + (unsigned)tid * 128;
#pragma unroll
    for (int k = 0; k < 8; ++k) async_b128(lb + k * 16, g + k * 16);
  }
  v8f racc[8];
#pragma unroll
  for (int n = 0; n < 8; ++n) {
    const float br = bres[n * 16 + chn];
#pragma unroll
    for (int j = 0; j < 8; ++j)
      racc[n][j] =
          bf2f(hin[(size_t)(t0 + j + (hiL ? 8 : 0)) * RES + n * 16 + chn]) + br;
  }
  wait_async0();
  __syncthreads();

  // ---- h_next = (h + z @ w_res + b_res) * sqrt(0.5)
#pragma unroll
  for (int n = 0; n < 8; ++n) {
#pragma unroll
    for (int kb = 0; kb < 4; ++kb) {
      v16bf b = *(const v16bf*)(wsh + (size_t)(kb * 8 + n) * 512 + lane * 16);
      racc[n] = wmma_bf16(zf[kb], b, racc[n]);
    }
#pragma unroll
    for (int j = 0; j < 8; ++j)
      hout[(size_t)(t0 + j + (hiL ? 8 : 0)) * RES + n * 16 + chn] =
          f2bf(racc[n][j] * SQRT_HALF);
  }
}

// o = relu(relu(skips*scale) @ w_out1 + b1) @ w_out2 + b2
__global__ __launch_bounds__(256) void wn_out(const float* __restrict__ skips,
                                              const __bf16* __restrict__ w1,
                                              const float* __restrict__ b1,
                                              const __bf16* __restrict__ w2,
                                              const float* __restrict__ b2,
                                              float* __restrict__ out) {
  __shared__ __bf16 obuf[8 * 16 * 128];
  const int lane = threadIdx.x & 31;
  const int wid  = threadIdx.x >> 5;
  const int t0   = (blockIdx.x * 8 + wid) * 16;
  const int m    = lane & 15;
  const bool hiL = lane >= 16;
  const int chn  = m;
  __bf16* ow = obuf + wid * 16 * 128;

  const float* srow = skips + (size_t)(t0 + m) * SKIPC;
  v16bf af[4];
#pragma unroll
  for (int kb = 0; kb < 4; ++kb) {
    const int cb = kb * 32 + (hiL ? 8 : 0);
    v16bf a;
#pragma unroll
    for (int i = 0; i < 8; ++i) {
      float v0 = srow[cb + i] * SKIP_SCALE;
      float v1 = srow[cb + 16 + i] * SKIP_SCALE;
      a[i]     = f2bf(v0 > 0.f ? v0 : 0.f);
      a[8 + i] = f2bf(v1 > 0.f ? v1 : 0.f);
    }
    af[kb] = a;
  }
#pragma unroll
  for (int n = 0; n < 8; ++n) {
    const float b0 = b1[n * 16 + chn];
    v8f acc;
#pragma unroll
    for (int j = 0; j < 8; ++j) acc[j] = b0;
#pragma unroll
    for (int kb = 0; kb < 4; ++kb) {
      v16bf b = *(const v16bf*)(w1 + ((size_t)(kb * 8 + n) * 32 + lane) * 16);
      acc = wmma_bf16(af[kb], b, acc);
    }
#pragma unroll
    for (int j = 0; j < 8; ++j) {
      float v = acc[j];
      ow[(j + (hiL ? 8 : 0)) * 128 + n * 16 + chn] = f2bf(v > 0.f ? v : 0.f);
    }
  }
  v16bf of[4];
#pragma unroll
  for (int kb = 0; kb < 4; ++kb) {
    const int cb = kb * 32 + (hiL ? 8 : 0);
    v8bf lo = *(const v8bf*)(ow + m * 128 + cb);
    v8bf hb = *(const v8bf*)(ow + m * 128 + cb + 16);
    v16bf a;
#pragma unroll
    for (int i = 0; i < 8; ++i) { a[i] = lo[i]; a[8 + i] = hb[i]; }
    of[kb] = a;
  }
#pragma unroll
  for (int n = 0; n < 16; ++n) {
    const float b0 = b2[n * 16 + chn];
    v8f acc;
#pragma unroll
    for (int j = 0; j < 8; ++j) acc[j] = b0;
#pragma unroll
    for (int kb = 0; kb < 4; ++kb) {
      v16bf b = *(const v16bf*)(w2 + ((size_t)(kb * 16 + n) * 32 + lane) * 16);
      acc = wmma_bf16(of[kb], b, acc);
    }
#pragma unroll
    for (int j = 0; j < 8; ++j)
      out[(size_t)(t0 + j + (hiL ? 8 : 0)) * OUTC + n * 16 + chn] = acc[j];
  }
}

// ------------------------------------------------------------------- host side

extern "C" void kernel_launch(void* const* d_in, const int* in_sizes, int n_in,
                              void* d_out, int out_size, void* d_ws,
                              size_t ws_size, hipStream_t stream) {
  const float* x       = (const float*)d_in[0];
  const float* c       = (const float*)d_in[1];
  const float* w_first = (const float*)d_in[2];
  const float* b_first = (const float*)d_in[3];
  const float* w_dil   = (const float*)d_in[4];
  const float* b_dil   = (const float*)d_in[5];
  const float* w_c     = (const float*)d_in[6];
  const float* w_skip  = (const float*)d_in[7];
  const float* b_skip  = (const float*)d_in[8];
  const float* w_res   = (const float*)d_in[9];
  const float* b_res   = (const float*)d_in[10];
  const float* w_out1  = (const float*)d_in[11];
  const float* b_out1  = (const float*)d_in[12];
  const float* w_out2  = (const float*)d_in[13];
  const float* b_out2  = (const float*)d_in[14];

  char* ws = (char*)d_ws;
  size_t off = 0;
  auto alloc = [&](size_t bytes) {
    size_t o = off;
    off += (bytes + 255) & ~(size_t)255;
    return o;
  };
  __bf16* h0 = (__bf16*)(ws + alloc((size_t)ROWS * RES * 2));
  __bf16* h1 = (__bf16*)(ws + alloc((size_t)ROWS * RES * 2));
  __bf16* cp = (__bf16*)(ws + alloc((size_t)ROWS * CPAD * 2));
  float* skp = (float*)(ws + alloc((size_t)ROWS * SKIPC * 4));
  __bf16* sw_first = (__bf16*)(ws + alloc((size_t)8 * 8 * 512 * 2));
  __bf16* sw_dil   = (__bf16*)(ws + alloc((size_t)20 * 16 * 12 * 512 * 2));
  __bf16* sw_c     = (__bf16*)(ws + alloc((size_t)20 * 16 * 3 * 512 * 2));
  __bf16* sw_skip  = (__bf16*)(ws + alloc((size_t)20 * 8 * 4 * 512 * 2));
  __bf16* sw_res   = (__bf16*)(ws + alloc((size_t)20 * 8 * 4 * 512 * 2));
  __bf16* sw_out1  = (__bf16*)(ws + alloc((size_t)8 * 4 * 512 * 2));
  __bf16* sw_out2  = (__bf16*)(ws + alloc((size_t)16 * 4 * 512 * 2));

  auto g = [](long n) { return (unsigned)((n + 255) / 256); };
  { long t = 1L * 8 * 8 * 512;    wn_swz<<<g(t), 256, 0, stream>>>(w_first, sw_first, 256, 8, 128, t); }
  { long t = 20L * 16 * 12 * 512; wn_swz<<<g(t), 256, 0, stream>>>(w_dil,   sw_dil,   384, 12, 256, t); }
  { long t = 20L * 16 * 3 * 512;  wn_swz<<<g(t), 256, 0, stream>>>(w_c,     sw_c,      80, 3, 256, t); }
  { long t = 20L * 8 * 4 * 512;   wn_swz<<<g(t), 256, 0, stream>>>(w_skip,  sw_skip,  128, 4, 128, t); }
  { long t = 20L * 8 * 4 * 512;   wn_swz<<<g(t), 256, 0, stream>>>(w_res,   sw_res,   128, 4, 128, t); }
  { long t = 1L * 8 * 4 * 512;    wn_swz<<<g(t), 256, 0, stream>>>(w_out1,  sw_out1,  128, 4, 128, t); }
  { long t = 1L * 16 * 4 * 512;   wn_swz<<<g(t), 256, 0, stream>>>(w_out2,  sw_out2,  128, 4, 256, t); }
  { long t = (long)ROWS * CPAD;   wn_cpad<<<g(t), 256, 0, stream>>>(c, cp, t); }
  { long t = (long)ROWS * SKIPC;  wn_zero<<<g(t), 256, 0, stream>>>(skp, t); }

  const unsigned NB = ROWS / 16 / 8;  // 128 blocks, 8 waves each
  wn_first<<<NB, 256, 0, stream>>>(x, sw_first, b_first, h0);
  for (int l = 0; l < 20; ++l) {
    int dil = 1 << (l % 10);
    const __bf16* hi_ = (l & 1) ? h1 : h0;
    __bf16* ho_       = (l & 1) ? h0 : h1;
    wn_layer<<<NB, 256, 0, stream>>>(
        hi_, ho_, cp, sw_dil + (long)l * 16 * 12 * 512,
        sw_c + (long)l * 16 * 3 * 512, sw_skip + (long)l * 8 * 4 * 512,
        sw_res + (long)l * 8 * 4 * 512, b_dil + l * 256, b_skip + l * 128,
        b_res + l * 128, skp, dil);
  }
  wn_out<<<NB, 256, 0, stream>>>(skp, sw_out1, b_out1, sw_out2, b_out2,
                                 (float*)d_out);
}
